// ScaledDotProductAttention_29042568855721
// MI455X (gfx1250) — matrix operand order repaired and checked by reading, not compiled, not run on hardware
//
#include <hip/hip_runtime.h>
#include <hip/hip_bf16.h>

// Problem constants (from reference)
#define B_    2
#define NH_   12
#define NQ_   2048
#define NK_   2048
#define D_    64

#define BQ    128       // query rows per workgroup (8 waves x 16)
#define BKT   64        // keys per iteration
#define NWAVE 8
#define NTHREADS (NWAVE * 32)
#define NITER (NK_ / BKT)

#define KSTRIDE 36      // uints per LDS row for K/V tiles (72 bf16, bank-padded)
#define PSTRIDE 34      // uints per LDS row for P tiles   (68 bf16, bank-padded)
#define BIASW   192     // staged bias diagonals per iteration (needs 191)

typedef __bf16 v16bf __attribute__((ext_vector_type(16)));
typedef float  v8f   __attribute__((ext_vector_type(8)));

union BF16x16 { unsigned u[8]; v16bf v; };

__device__ __forceinline__ unsigned pk_bf16(float a, float b) {
    unsigned ua = __float_as_uint(a);
    unsigned ub = __float_as_uint(b);
    ua = (ua + 0x7FFFu + ((ua >> 16) & 1u)) >> 16;
    ub = (ub + 0x7FFFu + ((ub >> 16) & 1u)) >> 16;
    return (ua & 0xFFFFu) | (ub << 16);
}
__device__ __forceinline__ unsigned short bf16_1(float a) {
    unsigned ua = __float_as_uint(a);
    return (unsigned short)((ua + 0x7FFFu + ((ua >> 16) & 1u)) >> 16);
}

// xor-butterfly within each 16-lane half using v_permlane16_b32 (pure VALU, no DS)
__device__ __forceinline__ float plx(float v, unsigned s0, unsigned s1) {
    return __uint_as_float(__builtin_amdgcn_permlane16(
        __float_as_uint(v), __float_as_uint(v), s0, s1, false, false));
}
__device__ __forceinline__ float rowmax16(float x) {
    x = fmaxf(x, plx(x, 0x67452301u, 0xEFCDAB89u));  // xor 1
    x = fmaxf(x, plx(x, 0x54761032u, 0xDCFE98BAu));  // xor 2
    x = fmaxf(x, plx(x, 0x32107654u, 0xBA98FEDCu));  // xor 4
    x = fmaxf(x, plx(x, 0xFEDCBA98u, 0x76543210u));  // xor 8
    return x;
}

__global__ __launch_bounds__(NTHREADS)
void fa_fwd_kernel(const float* __restrict__ qg, const float* __restrict__ kg,
                   const float* __restrict__ vg, const float* __restrict__ maskg,
                   const float* __restrict__ biasg, float* __restrict__ outg)
{
    __shared__ __attribute__((aligned(16))) unsigned kls[2][64 * KSTRIDE];
    __shared__ __attribute__((aligned(16))) unsigned vls[2][64 * KSTRIDE];
    __shared__ __attribute__((aligned(16))) unsigned pls[NWAVE * 16 * PSTRIDE];
    __shared__ __attribute__((aligned(16))) float    bias_s[2][BIASW];

    const int tid  = threadIdx.x;
    const int wave = tid >> 5;
    const int lane = tid & 31;
    const int ln   = lane & 15;   // N (or M for A-operand) index
    const int hh   = lane >> 4;   // half-wave select

    const int qblk = blockIdx.x;            // 0..NQ/BQ-1
    const int bh   = blockIdx.y;            // 0..B*NH-1
    const int b    = bh / NH_;
    const int h    = bh % NH_;

    const float* qh = qg + (size_t)bh * NQ_ * D_;
    const float* kh = kg + (size_t)bh * NK_ * D_;
    const float* vh = vg + (size_t)bh * NK_ * D_;
    const float* mh = maskg + (size_t)b * NQ_ * NK_;
    float*       oh = outg + (size_t)bh * NQ_ * D_;

    const int q0    = qblk * BQ;
    const int arow  = q0 + wave * 16 + ln;      // A-operand: M = lane&15
    const int qrow0 = q0 + wave * 16 + hh * 8;  // C/D layout base row for this lane

    // ---- Q -> bf16 A-operand registers (pre-scaled by 1/sqrt(D), exact pow2) ----
    BF16x16 aq[2];
    const float qscale = 0.125f;
#pragma unroll
    for (int c = 0; c < 2; ++c)
#pragma unroll
        for (int seg = 0; seg < 2; ++seg) {
            int dstart = 32 * c + 16 * hh + seg * 8;
            const float4* p = reinterpret_cast<const float4*>(qh + (size_t)arow * D_ + dstart);
            float4 f0 = p[0], f1 = p[1];
            aq[c].u[seg * 4 + 0] = pk_bf16(f0.x * qscale, f0.y * qscale);
            aq[c].u[seg * 4 + 1] = pk_bf16(f0.z * qscale, f0.w * qscale);
            aq[c].u[seg * 4 + 2] = pk_bf16(f1.x * qscale, f1.y * qscale);
            aq[c].u[seg * 4 + 3] = pk_bf16(f1.z * qscale, f1.w * qscale);
        }

    // all-ones bf16 B matrix for WMMA row sums
    BF16x16 ones;
#pragma unroll
    for (int i = 0; i < 8; ++i) ones.u[i] = 0x3F803F80u;

    // ---- online-softmax state (C/D layout: row = hh*8 + r, col = ln) ----
    float mrow[8], lrow[8];
    v8f o[4];
#pragma unroll
    for (int r = 0; r < 8; ++r) { mrow[r] = -INFINITY; lrow[r] = 0.0f; }
#pragma unroll
    for (int t = 0; t < 4; ++t) o[t] = (v8f){0.f,0.f,0.f,0.f,0.f,0.f,0.f,0.f};

    // ---- staging address setup (256 threads share one 64x64 K tile + V tile) ----
    const int skey = tid & 63;            // K staging: key row within tile
    const int sdh  = (tid >> 6) * 16;     // K staging: d quarter (16 floats)
    const int vkp  = tid & 31;            // V staging: key pair (keys 2*vkp, 2*vkp+1)
    const int vdq  = (tid >> 5) * 8;      // V staging: d group of 8
    const float* kptr = kh + (size_t)skey * D_ + sdh;
    const float* vptr = vh + (size_t)(2 * vkp) * D_ + vdq;
    // per-lane mask base pointer, advanced by BKT columns per iteration
    const float* mptr = mh + (size_t)qrow0 * NK_ + ln;
    // bias diagonal window base: diag index = q - k + NK, window starts at q0-kb-63+NK
    const float* bptr = biasg + (size_t)(q0 + NK_ - 63) * NH_ + h;  // advanced by -BKT*NH per iter

    for (int it = 0; it < NITER; ++it) {
        const int buf = it & 1;

        // ---- stage K (row-major bf16), V (transposed bf16), bias window ----
        {
            float4 kf[4];
#pragma unroll
            for (int i = 0; i < 4; ++i)
                kf[i] = reinterpret_cast<const float4*>(kptr)[i];
            uint4* kdst = reinterpret_cast<uint4*>(&kls[buf][skey * KSTRIDE + (sdh >> 1)]);
#pragma unroll
            for (int i = 0; i < 2; ++i) {
                uint4 u;
                u.x = pk_bf16(kf[2 * i].x,     kf[2 * i].y);
                u.y = pk_bf16(kf[2 * i].z,     kf[2 * i].w);
                u.z = pk_bf16(kf[2 * i + 1].x, kf[2 * i + 1].y);
                u.w = pk_bf16(kf[2 * i + 1].z, kf[2 * i + 1].w);
                kdst[i] = u;
            }
            float4 va[2], vb[2];
#pragma unroll
            for (int i = 0; i < 2; ++i) {
                va[i] = reinterpret_cast<const float4*>(vptr)[i];
                vb[i] = reinterpret_cast<const float4*>(vptr + D_)[i];
            }
            unsigned* vcol = &vls[buf][vkp];
#pragma unroll
            for (int i = 0; i < 2; ++i) {
                vcol[(vdq + 4 * i + 0) * KSTRIDE] = pk_bf16(va[i].x, vb[i].x);
                vcol[(vdq + 4 * i + 1) * KSTRIDE] = pk_bf16(va[i].y, vb[i].y);
                vcol[(vdq + 4 * i + 2) * KSTRIDE] = pk_bf16(va[i].z, vb[i].z);
                vcol[(vdq + 4 * i + 3) * KSTRIDE] = pk_bf16(va[i].w, vb[i].w);
            }
            if (tid < BIASW)
                bias_s[buf][tid] = bptr[(size_t)tid * NH_];
            kptr += (size_t)BKT * D_;
            vptr += (size_t)BKT * D_;
            bptr -= (size_t)BKT * NH_;
        }
        __syncthreads();

        // ---- S = Q*K^T : 8 WMMAs grouped (4 independent accumulator chains) ----
        v8f sv[4];
#pragma unroll
        for (int t = 0; t < 4; ++t) {
            sv[t] = (v8f){0.f,0.f,0.f,0.f,0.f,0.f,0.f,0.f};
#pragma unroll
            for (int c = 0; c < 2; ++c) {
                BF16x16 bk;
                const uint4* bp = reinterpret_cast<const uint4*>(
                    &kls[buf][(t * 16 + ln) * KSTRIDE + c * 16 + hh * 8]);
                uint4 u0 = bp[0], u1 = bp[1];
                bk.u[0] = u0.x; bk.u[1] = u0.y; bk.u[2] = u0.z; bk.u[3] = u0.w;
                bk.u[4] = u1.x; bk.u[5] = u1.y; bk.u[6] = u1.z; bk.u[7] = u1.w;
                sv[t] = __builtin_amdgcn_wmma_f32_16x16x32_bf16(
                        false, aq[c].v, false, bk.v, (short)0, sv[t], false, false);
            }
        }

        // ---- bias (LDS window, immediate offsets) + mask, added in place ----
#pragma unroll
        for (int t = 0; t < 4; ++t) {
            // diag idx within window: (q_local) - (k_local) + 63
            const float* bbase = &bias_s[buf][wave * 16 + hh * 8 - t * 16 - ln + 63];
#pragma unroll
            for (int r = 0; r < 8; ++r) {
                float bval = bbase[r];
                float mval = mptr[(size_t)r * NK_ + t * 16];
                sv[t][r] += (bval + mval);
            }
        }

        // ---- online softmax: permlane16 row-max, exp in place ----
        float alpha[8];
#pragma unroll
        for (int r = 0; r < 8; ++r) {
            float mx = fmaxf(fmaxf(sv[0][r], sv[1][r]), fmaxf(sv[2][r], sv[3][r]));
            mx = rowmax16(mx);
            float mnew = fmaxf(mrow[r], mx);
            alpha[r] = __expf(mrow[r] - mnew);
            mrow[r] = mnew;
#pragma unroll
            for (int t = 0; t < 4; ++t)
                sv[t][r] = __expf(sv[t][r] - mnew);
#pragma unroll
            for (int t = 0; t < 4; ++t)
                o[t][r] *= alpha[r];
        }

        // ---- P -> per-wave LDS (bf16), read back in A-operand layout ----
        {
            unsigned short* psh = reinterpret_cast<unsigned short*>(&pls[wave * 16 * PSTRIDE]);
#pragma unroll
            for (int r = 0; r < 8; ++r)
#pragma unroll
                for (int t = 0; t < 4; ++t)
                    psh[(hh * 8 + r) * (2 * PSTRIDE) + t * 16 + ln] = bf16_1(sv[t][r]);
        }
        BF16x16 pa[2];
        {
            const unsigned* prow = &pls[wave * 16 * PSTRIDE + ln * PSTRIDE];
#pragma unroll
            for (int c = 0; c < 2; ++c) {
                int j0 = 16 * c + hh * 8;
                const uint2* p0 = reinterpret_cast<const uint2*>(prow + j0);
                uint2 a0 = p0[0], a1 = p0[1];
                pa[c].u[0] = a0.x; pa[c].u[1] = a0.y; pa[c].u[2] = a1.x; pa[c].u[3] = a1.y;
                const uint2* p1 = reinterpret_cast<const uint2*>(prow + j0 + 4);
                uint2 b0 = p1[0], b1 = p1[1];
                pa[c].u[4] = b0.x; pa[c].u[5] = b0.y; pa[c].u[6] = b1.x; pa[c].u[7] = b1.y;
            }
        }

        // ---- row sums via WMMA against all-ones B: every column gets sum(P row) ----
        v8f sumacc = (v8f){0.f,0.f,0.f,0.f,0.f,0.f,0.f,0.f};
        sumacc = __builtin_amdgcn_wmma_f32_16x16x32_bf16(
                     false, pa[0].v, false, ones.v, (short)0, sumacc, false, false);
        sumacc = __builtin_amdgcn_wmma_f32_16x16x32_bf16(
                     false, pa[1].v, false, ones.v, (short)0, sumacc, false, false);
#pragma unroll
        for (int r = 0; r < 8; ++r)
            lrow[r] = lrow[r] * alpha[r] + sumacc[r];

        // ---- O += P*V (V^T tile from LDS in B-operand layout) ----
#pragma unroll
        for (int td = 0; td < 4; ++td) {
#pragma unroll
            for (int c = 0; c < 2; ++c) {
                BF16x16 bv;
                const uint4* vp4 = reinterpret_cast<const uint4*>(
                    &vls[buf][(td * 16 + ln) * KSTRIDE + c * 16 + hh * 8]);
                uint4 u0 = vp4[0], u1 = vp4[1];
                bv.u[0] = u0.x; bv.u[1] = u0.y; bv.u[2] = u0.z; bv.u[3] = u0.w;
                bv.u[4] = u1.x; bv.u[5] = u1.y; bv.u[6] = u1.z; bv.u[7] = u1.w;
                o[td] = __builtin_amdgcn_wmma_f32_16x16x32_bf16(
                            false, pa[c].v, false, bv.v, (short)0, o[td], false, false);
            }
        }

        mptr += BKT;
        // second barrier per iteration (after the one following staging): next iteration writes the other buffer,
        // whose previous readers all finished before this barrier.
        __syncthreads();
    }

    // ---- epilogue: divide by running sum, store f32 ----
#pragma unroll
    for (int r = 0; r < 8; ++r) {
        float inv = 1.0f / lrow[r];
        int qq = qrow0 + r;
#pragma unroll
        for (int td = 0; td < 4; ++td)
            oh[(size_t)qq * D_ + td * 16 + ln] = o[td][r] * inv;
    }
}

extern "C" void kernel_launch(void* const* d_in, const int* in_sizes, int n_in,
                              void* d_out, int out_size, void* d_ws, size_t ws_size,
                              hipStream_t stream) {
    const float* q    = (const float*)d_in[0];
    const float* k    = (const float*)d_in[1];
    const float* v    = (const float*)d_in[2];
    const float* mask = (const float*)d_in[3];
    const float* bias = (const float*)d_in[4];
    float* out        = (float*)d_out;
    (void)in_sizes; (void)n_in; (void)out_size; (void)d_ws; (void)ws_size;

    dim3 grid(NQ_ / BQ, B_ * NH_);
    fa_fwd_kernel<<<grid, NTHREADS, 0, stream>>>(q, k, v, mask, bias, out);
}
